// TorchMDEncoder_79714593014008
// MI455X (gfx1250) — matrix operand order repaired and checked by reading, not compiled, not run on hardware
//
#include <hip/hip_runtime.h>
#include <hip/hip_bf16.h>
#include <cstdint>
#include <cstddef>

// ---------------- problem constants (match reference) ----------------
#define N_ATOMS 100000
#define N_BLK   25000
#define N_GRAPH 64
#define N_EDGE  150000
#define HIDDIM  128
#define NHEADS  8
#define HDIM    16
#define NRBF    50
#define NLAYER  3
#define EATTR   16
#define CUT_R   5.0f

// padded row counts (multiples of 16) so GEMM needs no row masking
#define BP      25008
#define BP3     75008

typedef __attribute__((ext_vector_type(8)))  __bf16 v8bf;
typedef __attribute__((ext_vector_type(16))) __bf16 v16bf;
typedef __attribute__((ext_vector_type(8)))  float  v8f;

__device__ __forceinline__ float wave_sum(float v) {
#pragma unroll
  for (int m = 16; m > 0; m >>= 1) v += __shfl_xor(v, m, 32);
  return v;
}
__device__ __forceinline__ float siluf(float x) { return x / (1.0f + __expf(-x)); }

// ---------------- utility kernels ----------------
__global__ void fill0_kernel(float* __restrict__ p, size_t n) {
  size_t i = (size_t)blockIdx.x * blockDim.x + threadIdx.x;
  size_t st = (size_t)gridDim.x * blockDim.x;
  for (; i < n; i += st) p[i] = 0.0f;
}

__global__ void f32_to_bf16_kernel(const float* __restrict__ s, __bf16* __restrict__ d, size_t n) {
  size_t i = (size_t)blockIdx.x * blockDim.x + threadIdx.x;
  size_t st = (size_t)gridDim.x * blockDim.x;
  for (; i < n; i += st) d[i] = (__bf16)s[i];
}

// W [K, NC] fp32 row-major  ->  Wt [NC, Kpad] bf16 (zero-padded K)
__global__ void transpose_pad_kernel(const float* __restrict__ src, __bf16* __restrict__ dst,
                                     int K, int Kpad, int NC) {
  int id = blockIdx.x * blockDim.x + threadIdx.x;
  if (id >= NC * Kpad) return;
  int n = id / Kpad, kk = id % Kpad;
  dst[id] = (kk < K) ? (__bf16)src[kk * NC + n] : (__bf16)0.0f;
}

// ---------------- scatter_mean atoms -> blocks ----------------
__global__ void scatter_atoms_kernel(const float* __restrict__ H, const float* __restrict__ Z,
                                     const int* __restrict__ block_id,
                                     float* __restrict__ HbAcc, float* __restrict__ ZbAcc,
                                     float* __restrict__ cnt) {
  int a = blockIdx.x;
  int ch = threadIdx.x;
  int b = block_id[a];
  atomicAdd(&HbAcc[(size_t)b * HIDDIM + ch], H[(size_t)a * HIDDIM + ch]);
  if (ch < 3) atomicAdd(&ZbAcc[b * 3 + ch], Z[a * 3 + ch]);
  if (ch == 0) atomicAdd(&cnt[b], 1.0f);
}

__global__ void finalize_blocks_kernel(float* __restrict__ Hb, float* __restrict__ Zb,
                                       const float* __restrict__ cnt, float* __restrict__ x) {
  int b = blockIdx.x;
  int ch = threadIdx.x;
  float c = cnt[b];
  c = c < 1.0f ? 1.0f : c;
  float inv = 1.0f / c;
  float hv = Hb[(size_t)b * HIDDIM + ch] * inv;
  Hb[(size_t)b * HIDDIM + ch] = hv;      // Hb output (d_out region 0)
  x[(size_t)b * HIDDIM + ch] = hv;       // layer state
  if (ch < 3) Zb[b * 3 + ch] *= inv;
}

// ---------------- edge geometry + RBF + edge linear ----------------
__global__ void edge_geom_kernel(const int* __restrict__ edges, const float* __restrict__ Zb,
                                 const float* __restrict__ edge_attr,
                                 const float* __restrict__ W_edge, const float* __restrict__ b_edge,
                                 const float* __restrict__ means, const float* __restrict__ betas,
                                 float* __restrict__ cutfm, float* __restrict__ fmA,
                                 float* __restrict__ dij, __bf16* __restrict__ fij) {
  int e = blockIdx.x * blockDim.x + threadIdx.x;
  if (e >= N_EDGE) return;
  int s = edges[e], d = edges[N_EDGE + e];
  float ex = Zb[s * 3 + 0] - Zb[d * 3 + 0];
  float ey = Zb[s * 3 + 1] - Zb[d * 3 + 1];
  float ez = Zb[s * 3 + 2] - Zb[d * 3 + 2];
  float r = sqrtf(ex * ex + ey * ey + ez * ez + 1e-12f);
  float fm = (r > 0.01f) ? 1.0f : 0.0f;
  float invr = fm / r;
  dij[e * 3 + 0] = ex * invr;
  dij[e * 3 + 1] = ey * invr;
  dij[e * 3 + 2] = ez * invr;
  float cut = (r < CUT_R) ? 0.5f * (cosf(3.14159265358979f * r / CUT_R) + 1.0f) : 0.0f;
  cutfm[e] = cut * fm;
  fmA[e] = fm;
  float er = expf(-r);
  float ea[EATTR];
#pragma unroll
  for (int k2 = 0; k2 < EATTR; ++k2) ea[k2] = edge_attr[(size_t)e * EATTR + k2];
  for (int j = 0; j < NRBF; ++j) {
    float t = er - means[j];
    float rbf = cut * expf(-betas[j] * t * t);
    float lin = b_edge[j];
#pragma unroll
    for (int k2 = 0; k2 < EATTR; ++k2) lin += ea[k2] * W_edge[k2 * NRBF + j];
    fij[(size_t)e * 64 + j] = (__bf16)(rbf + lin);
  }
  for (int j = NRBF; j < 64; ++j) fij[(size_t)e * 64 + j] = (__bf16)0.0f;
}

// ---------------- layernorm -> bf16 (one wave per 128-wide row) ----------------
__global__ void ln_bf16_kernel(const float* __restrict__ X, const float* __restrict__ w,
                               const float* __restrict__ b, __bf16* __restrict__ out, int rows) {
  int wid = threadIdx.x >> 5, lane = threadIdx.x & 31;
  int row = blockIdx.x * 8 + wid;
  if (row >= rows) return;
  float vals[4], s = 0.0f, ss = 0.0f;
#pragma unroll
  for (int i = 0; i < 4; ++i) {
    float t = X[(size_t)row * HIDDIM + lane + 32 * i];
    vals[i] = t; s += t; ss += t * t;
  }
  s = wave_sum(s); ss = wave_sum(ss);
  float mean = s * (1.0f / HIDDIM);
  float var = ss * (1.0f / HIDDIM) - mean * mean;
  float inv = rsqrtf(var + 1e-5f);
#pragma unroll
  for (int i = 0; i < 4; ++i) {
    int c = lane + 32 * i;
    out[(size_t)row * HIDDIM + c] = (__bf16)((vals[i] - mean) * inv * w[c] + b[c]);
  }
}

// ---------------- WMMA bf16 GEMM: out[Mp,NC] = X[Mp,K] @ Wt[NC,K]^T + bias ----------------
// Rows pre-padded to a multiple of 16 -> no row masking.
// One wave owns a 16-row tile; A regs held across the full N sweep.
// B tiles ping-pong between two register sets (ct loop unrolled x2, colTiles is even),
// so no inter-buffer register copies and loads overlap the WMMA pipe.
template <int KSTEPS, int ACT, int OUTB>
__global__ __launch_bounds__(256) void gemm_bf16_kernel(
    const __bf16* __restrict__ X, const __bf16* __restrict__ Wt,
    const float* __restrict__ bias,  // always non-null (zeros if no bias)
    float* __restrict__ outF, __bf16* __restrict__ outH,
    int NC, int rowTiles) {
  constexpr int K = KSTEPS * 32;
  const int rt = blockIdx.x * 8 + (threadIdx.x >> 5);
  if (rt >= rowTiles) return;
  const int lane = threadIdx.x & 31;
  const int mi = lane & 15;
  const int hh = lane >> 4;
  const int row = rt * 16 + mi;
  const int n = lane & 15;

  // A tile layout (16-bit A 16x32, ISA 7.12.2): lane-half hh selects K phase.
  v16bf A[KSTEPS];
#pragma unroll
  for (int ks = 0; ks < KSTEPS; ++ks) {
    const __bf16* xp = X + (size_t)row * K + ks * 32 + hh * 8;
    v8bf a0 = *(const v8bf*)(xp);        // K = 32ks + 8h .. +7
    v8bf a1 = *(const v8bf*)(xp + 16);   // K = 32ks + 16 + 8h .. +7
    A[ks] = __builtin_shufflevector(a0, a1, 0, 1, 2, 3, 4, 5, 6, 7, 8, 9, 10, 11, 12, 13, 14, 15);
  }

  const int colTiles = NC >> 4;  // 8 or 24: always even
  // B tile: column = lane&15; K = 32ks + 8h .. +7 and 32ks + 16 + 8h .. +7 (Wt is [NC,K])
  auto loadB = [&](int ct, v16bf* Bv) {
#pragma unroll
    for (int ks = 0; ks < KSTEPS; ++ks) {
      const __bf16* wp = Wt + (size_t)(ct * 16 + n) * K + ks * 32 + hh * 8;
      v8bf b0 = *(const v8bf*)(wp);
      v8bf b1 = *(const v8bf*)(wp + 16);
      Bv[ks] = __builtin_shufflevector(b0, b1, 0, 1, 2, 3, 4, 5, 6, 7, 8, 9, 10, 11, 12, 13, 14, 15);
    }
  };
  auto compute = [&](int ct, const v16bf* Bv) {
    const int col = ct * 16 + n;
    float bval = bias[col];
    v8f c = {0.f, 0.f, 0.f, 0.f, 0.f, 0.f, 0.f, 0.f};
#pragma unroll
    for (int ks = 0; ks < KSTEPS; ++ks)
      c = __builtin_amdgcn_wmma_f32_16x16x32_bf16(false, A[ks], false, Bv[ks], (short)0, c,
                                                  false, false);
#pragma unroll
    for (int r = 0; r < 8; ++r) {
      int ro = rt * 16 + r + hh * 8;   // C/D layout: vgpr r -> M = r + 8*half
      float val = c[r] + bval;
      if (ACT == 1) val = siluf(val);
      if (OUTB) outH[(size_t)ro * NC + col] = (__bf16)val;
      else      outF[(size_t)ro * NC + col] = val;
    }
  };

  v16bf B0[KSTEPS], B1[KSTEPS];
  loadB(0, B0);
  for (int ct = 0; ct < colTiles; ct += 2) {
    loadB(ct + 1, B1);
    compute(ct, B0);
    int ctn = (ct + 2 < colTiles) ? ct + 2 : 0;  // clamp to a valid tile
    loadB(ctn, B0);
    compute(ct + 1, B1);
  }
}

// ---------------- edge message + scatter (one block per edge, ch-parallel) ----------------
__global__ void edge_msg_kernel(const int* __restrict__ edges,
                                const float* __restrict__ q, const float* __restrict__ k,
                                const __bf16* __restrict__ dkb,
                                const float* __restrict__ v, const __bf16* __restrict__ dvb,
                                const float* __restrict__ vec, const float* __restrict__ dij,
                                const float* __restrict__ cutfm, const float* __restrict__ fmA,
                                float* __restrict__ msum, float* __restrict__ vecmsum) {
  int e = blockIdx.x;
  int ch = threadIdx.x;  // 0..127
  int src = edges[e], dst = edges[N_EDGE + e];
  __shared__ float sprod[HIDDIM];
  __shared__ float sattn[NHEADS];
  float qv = q[(size_t)dst * HIDDIM + ch];
  float kv = k[(size_t)src * HIDDIM + ch];
  float dkv = (float)dkb[(size_t)e * HIDDIM + ch];
  sprod[ch] = qv * kv * dkv;
  __syncthreads();
  if (ch < NHEADS) {
    float s = 0.0f;
#pragma unroll
    for (int d2 = 0; d2 < HDIM; ++d2) s += sprod[ch * HDIM + d2];
    sattn[ch] = siluf(s) * cutfm[e];
  }
  __syncthreads();
  int h = ch >> 4;
  float attn = sattn[h];
  size_t vb = (size_t)src * 384 + (ch & 15) + h * 48;
  size_t db = (size_t)e * 384 + (ch & 15) + h * 48;
  float vx = v[vb], v1 = v[vb + 16], v2 = v[vb + 32];
  float dvx = (float)dvb[db], dv1 = (float)dvb[db + 16], dv2 = (float)dvb[db + 32];
  float xm = vx * dvx * attn;
  atomicAdd(&msum[(size_t)dst * HIDDIM + ch], xm);
  float w1 = v1 * dv1, w2 = v2 * dv2;
  float fmv = fmA[e];
#pragma unroll
  for (int c3 = 0; c3 < 3; ++c3) {
    float vm = (vec[((size_t)src * 3 + c3) * HIDDIM + ch] * w1 + w2 * dij[e * 3 + c3]) * fmv;
    atomicAdd(&vecmsum[((size_t)dst * 3 + c3) * HIDDIM + ch], vm);
  }
}

// ---------------- per-layer state update ----------------
__global__ void update_kernel(float* __restrict__ x, float* __restrict__ vec,
                              const float* __restrict__ vecW, const float* __restrict__ o,
                              const float* __restrict__ vecmsum) {
  int idx = blockIdx.x * blockDim.x + threadIdx.x;
  if (idx >= N_BLK * HIDDIM) return;
  int b = idx >> 7, ch = idx & 127;
  float vdot = 0.0f;
#pragma unroll
  for (int c3 = 0; c3 < 3; ++c3) {
    size_t rb = ((size_t)b * 3 + c3) * 384;
    vdot += vecW[rb + ch] * vecW[rb + 128 + ch];
  }
  size_t ob = (size_t)b * 384;
  float o1 = o[ob + ch], o2 = o[ob + 128 + ch], o3 = o[ob + 256 + ch];
  x[idx] += vdot * o2 + o3;
#pragma unroll
  for (int c3 = 0; c3 < 3; ++c3) {
    size_t vr = ((size_t)b * 3 + c3) * HIDDIM + ch;
    vec[vr] += vecW[((size_t)b * 3 + c3) * 384 + 256 + ch] * o1 + vecmsum[vr];
  }
}

// ---------------- final block layernorm + L2 norm + graph scatter ----------------
__global__ void final_block_kernel(const float* __restrict__ X, const float* __restrict__ w,
                                   const float* __restrict__ b, const int* __restrict__ batch_id,
                                   float* __restrict__ outBR, float* __restrict__ gacc) {
  int wid = threadIdx.x >> 5, lane = threadIdx.x & 31;
  int row = blockIdx.x * 8 + wid;
  if (row >= N_BLK) return;
  float vals[4], s = 0.0f, ss = 0.0f;
#pragma unroll
  for (int i = 0; i < 4; ++i) {
    float t = X[(size_t)row * HIDDIM + lane + 32 * i];
    vals[i] = t; s += t; ss += t * t;
  }
  s = wave_sum(s); ss = wave_sum(ss);
  float mean = s * (1.0f / HIDDIM);
  float var = ss * (1.0f / HIDDIM) - mean * mean;
  float inv = rsqrtf(var + 1e-5f);
  float t4[4], s2 = 0.0f;
#pragma unroll
  for (int i = 0; i < 4; ++i) {
    int c = lane + 32 * i;
    t4[i] = (vals[i] - mean) * inv * w[c] + b[c];
    s2 += t4[i] * t4[i];
  }
  s2 = wave_sum(s2);
  float nrm = fmaxf(sqrtf(s2), 1e-12f);
  float invn = 1.0f / nrm;
  int g = batch_id[row];
#pragma unroll
  for (int i = 0; i < 4; ++i) {
    int c = lane + 32 * i;
    float br = t4[i] * invn;
    outBR[(size_t)row * HIDDIM + c] = br;
    atomicAdd(&gacc[(size_t)g * HIDDIM + c], br);
  }
}

__global__ void graph_norm_kernel(const float* __restrict__ gacc, float* __restrict__ out) {
  int wid = threadIdx.x >> 5, lane = threadIdx.x & 31;
  int row = blockIdx.x * 8 + wid;
  if (row >= N_GRAPH) return;
  float vals[4], ss = 0.0f;
#pragma unroll
  for (int i = 0; i < 4; ++i) {
    float t = gacc[(size_t)row * HIDDIM + lane + 32 * i];
    vals[i] = t; ss += t * t;
  }
  ss = wave_sum(ss);
  float nrm = fmaxf(sqrtf(ss), 1e-12f);
  float invn = 1.0f / nrm;
#pragma unroll
  for (int i = 0; i < 4; ++i)
    out[(size_t)row * HIDDIM + lane + 32 * i] = vals[i] * invn;
}

// ==================================================================
extern "C" void kernel_launch(void* const* d_in, const int* in_sizes, int n_in,
                              void* d_out, int out_size, void* d_ws, size_t ws_size,
                              hipStream_t stream) {
  (void)in_sizes; (void)n_in; (void)out_size; (void)ws_size;
  const float* H         = (const float*)d_in[0];
  const float* Z         = (const float*)d_in[1];
  const float* edge_attr = (const float*)d_in[2];
  const float* W_edge    = (const float*)d_in[3];
  const float* b_edge    = (const float*)d_in[4];
  const float* rbf_means = (const float*)d_in[5];
  const float* rbf_betas = (const float*)d_in[6];
  const float* ln_w      = (const float*)d_in[7];
  const float* ln_b      = (const float*)d_in[8];
  const float* Wq        = (const float*)d_in[9];
  const float* bq        = (const float*)d_in[10];
  const float* Wk        = (const float*)d_in[11];
  const float* bk        = (const float*)d_in[12];
  const float* Wv        = (const float*)d_in[13];
  const float* bv        = (const float*)d_in[14];
  const float* Wo        = (const float*)d_in[15];
  const float* bo        = (const float*)d_in[16];
  const float* Wvec      = (const float*)d_in[17];
  const float* Wdk       = (const float*)d_in[18];
  const float* bdk       = (const float*)d_in[19];
  const float* Wdv       = (const float*)d_in[20];
  const float* bdv       = (const float*)d_in[21];
  const float* out_w     = (const float*)d_in[22];
  const float* out_b     = (const float*)d_in[23];
  const int* block_id    = (const int*)d_in[24];
  const int* batch_id    = (const int*)d_in[25];
  const int* edges       = (const int*)d_in[26];

  float* out0 = (float*)d_out;                       // Hb        [B,128]
  float* out1 = out0 + (size_t)N_BLK * HIDDIM;       // block_repr[B,128]
  float* out2 = out1 + (size_t)N_BLK * HIDDIM;       // graph_repr[G,128]

  // ---- workspace carve-out (GEMM-facing buffers use padded rows) ----
  char* wsb = (char*)d_ws;
  size_t off = 0;
  auto AL = [&](size_t bytes) -> void* {
    void* p = wsb + off;
    off += (bytes + 255) & ~(size_t)255;
    return p;
  };
  float*  cnt     = (float*)AL((size_t)N_BLK * 4);
  float*  Zb      = (float*)AL((size_t)N_BLK * 3 * 4);
  float*  x       = (float*)AL((size_t)N_BLK * HIDDIM * 4);
  float*  vec     = (float*)AL((size_t)N_BLK * 3 * HIDDIM * 4);
  float*  cutfm   = (float*)AL((size_t)N_EDGE * 4);
  float*  fmA     = (float*)AL((size_t)N_EDGE * 4);
  float*  dij     = (float*)AL((size_t)N_EDGE * 3 * 4);
  __bf16* fij     = (__bf16*)AL((size_t)N_EDGE * 64 * 2);
  // transposed bf16 weights, per layer: q,k(128x128) v,o,vec(384x128) dk(128x64) dv(384x64)
  const size_t OFF_Q = 0, OFF_K = 16384, OFF_V = 32768, OFF_O = 81920,
               OFF_VEC = 131072, OFF_DK = 180224, OFF_DV = 188416, WPL = 212992;
  __bf16* wT      = (__bf16*)AL((size_t)NLAYER * WPL * 2);
  __bf16* xn      = (__bf16*)AL((size_t)BP * HIDDIM * 2);
  __bf16* vecbf   = (__bf16*)AL((size_t)BP3 * HIDDIM * 2);
  float*  qbuf    = (float*)AL((size_t)BP * HIDDIM * 4);
  float*  kbuf    = (float*)AL((size_t)BP * HIDDIM * 4);
  float*  vbuf    = (float*)AL((size_t)BP * 384 * 4);
  float*  vecW    = (float*)AL((size_t)BP3 * 384 * 4);
  __bf16* dk      = (__bf16*)AL((size_t)N_EDGE * HIDDIM * 2);
  __bf16* dv      = (__bf16*)AL((size_t)N_EDGE * 384 * 2);
  float*  msum    = (float*)AL((size_t)N_BLK * HIDDIM * 4);
  __bf16* msumbf  = (__bf16*)AL((size_t)BP * HIDDIM * 2);
  float*  vecmsum = (float*)AL((size_t)N_BLK * 3 * HIDDIM * 4);
  float*  obuf    = (float*)AL((size_t)BP * 384 * 4);
  float*  gacc    = (float*)AL((size_t)N_GRAPH * HIDDIM * 4);
  float*  zbias   = (float*)AL((size_t)384 * 4);

  auto zero = [&](float* p, size_t n) {
    unsigned g = (unsigned)((n + 255) / 256);
    if (g > 8192) g = 8192;
    fill0_kernel<<<g, 256, 0, stream>>>(p, n);
  };

  const int RT_B  = BP / 16;      // 1563
  const int RT_3B = BP3 / 16;     // 4688
  const int RT_E  = N_EDGE / 16;  // 9375

  // ---- init ----
  zero(out0, (size_t)N_BLK * HIDDIM);
  zero(cnt, N_BLK);
  zero(Zb, (size_t)N_BLK * 3);
  zero(vec, (size_t)N_BLK * 3 * HIDDIM);
  zero(gacc, (size_t)N_GRAPH * HIDDIM);
  zero(zbias, 384);

  // ---- weight transposes (bf16) ----
  for (int l = 0; l < NLAYER; ++l) {
    __bf16* wl = wT + (size_t)l * WPL;
    auto tp = [&](const float* src, __bf16* dst, int K, int Kpad, int NC) {
      int n = NC * Kpad;
      transpose_pad_kernel<<<(n + 255) / 256, 256, 0, stream>>>(src, dst, K, Kpad, NC);
    };
    tp(Wq   + (size_t)l * 128 * 128, wl + OFF_Q,   128, 128, 128);
    tp(Wk   + (size_t)l * 128 * 128, wl + OFF_K,   128, 128, 128);
    tp(Wv   + (size_t)l * 128 * 384, wl + OFF_V,   128, 128, 384);
    tp(Wo   + (size_t)l * 128 * 384, wl + OFF_O,   128, 128, 384);
    tp(Wvec + (size_t)l * 128 * 384, wl + OFF_VEC, 128, 128, 384);
    tp(Wdk  + (size_t)l * NRBF * 128, wl + OFF_DK, NRBF, 64, 128);
    tp(Wdv  + (size_t)l * NRBF * 384, wl + OFF_DV, NRBF, 64, 384);
  }

  // ---- scatter mean ----
  scatter_atoms_kernel<<<N_ATOMS, 128, 0, stream>>>(H, Z, block_id, out0, Zb, cnt);
  finalize_blocks_kernel<<<N_BLK, 128, 0, stream>>>(out0, Zb, cnt, x);

  // ---- edge geometry ----
  edge_geom_kernel<<<(N_EDGE + 127) / 128, 128, 0, stream>>>(
      edges, Zb, edge_attr, W_edge, b_edge, rbf_means, rbf_betas, cutfm, fmA, dij, fij);

  // ---- layers ----
  for (int l = 0; l < NLAYER; ++l) {
    const __bf16* wl = wT + (size_t)l * WPL;
    ln_bf16_kernel<<<(N_BLK + 7) / 8, 256, 0, stream>>>(x, ln_w + l * HIDDIM, ln_b + l * HIDDIM,
                                                        xn, N_BLK);
    f32_to_bf16_kernel<<<8192, 256, 0, stream>>>(vec, vecbf, (size_t)N_BLK * 3 * HIDDIM);

    gemm_bf16_kernel<4, 0, 0><<<(RT_B + 7) / 8, 256, 0, stream>>>(
        xn, wl + OFF_Q, bq + l * 128, qbuf, nullptr, 128, RT_B);
    gemm_bf16_kernel<4, 0, 0><<<(RT_B + 7) / 8, 256, 0, stream>>>(
        xn, wl + OFF_K, bk + l * 128, kbuf, nullptr, 128, RT_B);
    gemm_bf16_kernel<4, 0, 0><<<(RT_B + 7) / 8, 256, 0, stream>>>(
        xn, wl + OFF_V, bv + l * 384, vbuf, nullptr, 384, RT_B);
    gemm_bf16_kernel<4, 0, 0><<<(RT_3B + 7) / 8, 256, 0, stream>>>(
        vecbf, wl + OFF_VEC, zbias, vecW, nullptr, 384, RT_3B);
    gemm_bf16_kernel<2, 1, 1><<<(RT_E + 7) / 8, 256, 0, stream>>>(
        fij, wl + OFF_DK, bdk + l * 128, nullptr, dk, 128, RT_E);
    gemm_bf16_kernel<2, 1, 1><<<(RT_E + 7) / 8, 256, 0, stream>>>(
        fij, wl + OFF_DV, bdv + l * 384, nullptr, dv, 384, RT_E);

    zero(msum, (size_t)N_BLK * HIDDIM);
    zero(vecmsum, (size_t)N_BLK * 3 * HIDDIM);

    edge_msg_kernel<<<N_EDGE, 128, 0, stream>>>(edges, qbuf, kbuf, dk, vbuf, dv,
                                                vec, dij, cutfm, fmA, msum, vecmsum);

    f32_to_bf16_kernel<<<8192, 256, 0, stream>>>(msum, msumbf, (size_t)N_BLK * HIDDIM);
    gemm_bf16_kernel<4, 0, 0><<<(RT_B + 7) / 8, 256, 0, stream>>>(
        msumbf, wl + OFF_O, bo + l * 384, obuf, nullptr, 384, RT_B);

    update_kernel<<<(N_BLK * HIDDIM + 255) / 256, 256, 0, stream>>>(x, vec, vecW, obuf, vecmsum);
  }

  // ---- final norms + graph pooling ----
  final_block_kernel<<<(N_BLK + 7) / 8, 256, 0, stream>>>(x, out_w, out_b, batch_id, out1, gacc);
  graph_norm_kernel<<<(N_GRAPH + 7) / 8, 256, 0, stream>>>(gacc, out2);
}
